// ResidualVQ_2413771620473
// MI455X (gfx1250) — matrix operand order repaired and checked by reading, not compiled, not run on hardware
//
#include <hip/hip_runtime.h>
#include <hip/hip_fp16.h>

typedef __attribute__((ext_vector_type(16))) _Float16 v16h;
typedef __attribute__((ext_vector_type(8)))  float    v8f;

#define B_  16
#define N_  2048
#define D_  512
#define Q_  8
#define C_  1024
#define M_  (B_ * N_)          // 32768 rows
#define EPS 1e-6f

// ---------------------------------------------------------------------------
// helpers
// ---------------------------------------------------------------------------
__device__ __forceinline__ unsigned long long shfl_xor_u64(unsigned long long v, int mask) {
    int lo = (int)(unsigned)(v & 0xFFFFFFFFull);
    int hi = (int)(unsigned)(v >> 32);
    lo = __shfl_xor(lo, mask, 32);
    hi = __shfl_xor(hi, mask, 32);
    return ((unsigned long long)(unsigned)hi << 32) | (unsigned)lo;
}

__device__ __forceinline__ unsigned long long u64min(unsigned long long a, unsigned long long b) {
    return a < b ? a : b;
}

// monotone float -> uint mapping (preserves order, works for +/- values)
__device__ __forceinline__ unsigned ordered_bits(float f) {
    unsigned u = __float_as_uint(f);
    return (u & 0x80000000u) ? ~u : (u | 0x80000000u);
}

// ---------------------------------------------------------------------------
// kernel 0a: codebooks -> hi/lo f16 split + ||c||^2 (fp32, exact)
// one 256-thread block per codebook row (Q_*C_ rows)
// ---------------------------------------------------------------------------
__global__ void k_init_cb(const float* __restrict__ cb,
                          _Float16* __restrict__ cb_hi, _Float16* __restrict__ cb_lo,
                          float* __restrict__ csq) {
    int row = blockIdx.x;                       // 0 .. Q_*C_-1
    const float* src = cb + (size_t)row * D_;
    float ss = 0.f;
    for (int i = threadIdx.x; i < D_; i += 256) {
        float v = src[i];
        _Float16 h = (_Float16)v;
        cb_hi[(size_t)row * D_ + i] = h;
        cb_lo[(size_t)row * D_ + i] = (_Float16)(v - (float)h);
        ss += v * v;
    }
    __shared__ float s[256];
    s[threadIdx.x] = ss;
    __syncthreads();
    for (int st = 128; st > 0; st >>= 1) {
        if (threadIdx.x < st) s[threadIdx.x] += s[threadIdx.x + st];
        __syncthreads();
    }
    if (threadIdx.x == 0) csq[row] = s[0];
}

// ---------------------------------------------------------------------------
// kernel 0b: residual = x (fp32 + hi/lo f16), quantized_out = 0
// ---------------------------------------------------------------------------
__global__ void k_init_x(const float* __restrict__ x,
                         float* __restrict__ resid,
                         _Float16* __restrict__ rhi, _Float16* __restrict__ rlo,
                         float* __restrict__ outq) {
    size_t total = (size_t)M_ * D_;
    for (size_t i = (size_t)blockIdx.x * blockDim.x + threadIdx.x; i < total;
         i += (size_t)gridDim.x * blockDim.x) {
        float v = x[i];
        resid[i] = v;
        _Float16 h = (_Float16)v;
        rhi[i] = h;
        rlo[i] = (_Float16)(v - (float)h);
        outq[i] = 0.f;
    }
}

// ---------------------------------------------------------------------------
// kernel 1a: reset per-row argmin keys
// ---------------------------------------------------------------------------
__global__ void k_reset_argmin(unsigned long long* __restrict__ rowmin) {
    int i = blockIdx.x * blockDim.x + threadIdx.x;
    if (i < M_) rowmin[i] = ~0ull;
}

// ---------------------------------------------------------------------------
// kernel 1b: distance GEMM (split-f16 WMMA) + per-row argmin
//   Each wave computes a 16(M) x 128(C) strip; 4 waves / block.
//   score(m,c) = csq[c] - 2 * (r . c)   (||r||^2 is row-constant, argmin-safe)
// ---------------------------------------------------------------------------
__global__ __launch_bounds__(128) void k_dist_argmin(
        const _Float16* __restrict__ rhi, const _Float16* __restrict__ rlo,
        const _Float16* __restrict__ cbhi, const _Float16* __restrict__ cblo,
        const float* __restrict__ csq,
        unsigned long long* __restrict__ rowmin, int qi) {
    const int wave  = threadIdx.x >> 5;
    const int lane  = threadIdx.x & 31;
    const int strip = blockIdx.x * 4 + wave;      // 0 .. 16383
    const int mtile = strip >> 3;                 // 0 .. 2047
    const int ngrp  = strip & 7;                  // 0 .. 7
    const int m0 = mtile * 16;
    const int c0 = ngrp * 128;
    const int g   = lane >> 4;                    // lane-half
    const int l16 = lane & 15;

    const _Float16* cb_h = cbhi + (size_t)qi * C_ * D_;
    const _Float16* cb_l = cblo + (size_t)qi * C_ * D_;
    const float*    cs   = csq + qi * C_;

    v8f acc[8];
#pragma unroll
    for (int nt = 0; nt < 8; ++nt) acc[nt] = (v8f){0.f,0.f,0.f,0.f,0.f,0.f,0.f,0.f};

    // A-row pointers for this lane (16-bit A 16x32 layout: lane holds row m0+l16)
    const unsigned* arow_h = (const unsigned*)(rhi + (size_t)(m0 + l16) * D_);
    const unsigned* arow_l = (const unsigned*)(rlo + (size_t)(m0 + l16) * D_);

    for (int kc = 0; kc < D_; kc += 32) {
        // ---- A fragments (hi & lo) ----
        union { v16h v; unsigned u[8]; } Ah, Al;
#pragma unroll
        for (int r = 0; r < 8; ++r) {
            int k = kc + ((r < 4) ? (g * 8 + 2 * r) : (16 + g * 8 + 2 * (r - 4)));
            Ah.u[r] = arow_h[k >> 1];
            Al.u[r] = arow_l[k >> 1];
        }
        if (kc + 32 < D_) {
            __builtin_prefetch(arow_h + ((kc + 32) >> 1), 0, 1);
            __builtin_prefetch(arow_l + ((kc + 32) >> 1), 0, 1);
        }
        // ---- loop over 8 column tiles; 3 WMMAs each (hi*hi + hi*lo + lo*hi) ----
#pragma unroll
        for (int nt = 0; nt < 8; ++nt) {
            int col = c0 + nt * 16 + l16;         // B lane holds column `col`
            const unsigned* bh = (const unsigned*)(cb_h + (size_t)col * D_);
            const unsigned* bl = (const unsigned*)(cb_l + (size_t)col * D_);
            union { v16h v; unsigned u[8]; } Bh, Bl;
#pragma unroll
            for (int r = 0; r < 8; ++r) {
                int k = kc + ((r < 4) ? (g * 8 + 2 * r) : (16 + g * 8 + 2 * (r - 4)));      // B 32x16 layout: same k order as A
                Bh.u[r] = bh[k >> 1];
                Bl.u[r] = bl[k >> 1];
            }
            acc[nt] = __builtin_amdgcn_wmma_f32_16x16x32_f16(false, Ah.v, false, Bh.v,
                                                             (short)0, acc[nt], false, false);
            acc[nt] = __builtin_amdgcn_wmma_f32_16x16x32_f16(false, Ah.v, false, Bl.v,
                                                             (short)0, acc[nt], false, false);
            acc[nt] = __builtin_amdgcn_wmma_f32_16x16x32_f16(false, Al.v, false, Bh.v,
                                                             (short)0, acc[nt], false, false);
        }
    }

    // ---- per-row argmin over this wave's 128 columns, then global atomicMin ----
    // C/D layout: VGPR j, lane: row = m0 + j + g*8, col = c0 + nt*16 + l16
#pragma unroll
    for (int j = 0; j < 8; ++j) {
        unsigned long long best = ~0ull;
#pragma unroll
        for (int nt = 0; nt < 8; ++nt) {
            int c = c0 + nt * 16 + l16;
            float score = cs[c] - 2.0f * acc[nt][j];
            unsigned long long key =
                ((unsigned long long)ordered_bits(score) << 32) | (unsigned)c;
            best = u64min(best, key);
        }
        // reduce across the 16 lanes of this lane-half
        best = u64min(best, shfl_xor_u64(best, 1));
        best = u64min(best, shfl_xor_u64(best, 2));
        best = u64min(best, shfl_xor_u64(best, 4));
        best = u64min(best, shfl_xor_u64(best, 8));
        if (l16 == 0) {
            int row = m0 + j + g * 8;
            atomicMin(rowmin + row, best);
        }
    }
}

// ---------------------------------------------------------------------------
// kernel 2: gather codebook row, rotation-trick transform, residual update
// one 256-thread block per row; each thread owns elements t and t+256
// ---------------------------------------------------------------------------
__global__ void k_rotate_update(const float* __restrict__ cb,
                                const unsigned long long* __restrict__ rowmin,
                                float* __restrict__ resid,
                                _Float16* __restrict__ rhi, _Float16* __restrict__ rlo,
                                float* __restrict__ outq,
                                float* __restrict__ loss_rows,
                                float* __restrict__ out_idx, int qi) {
    const int m = blockIdx.x;
    const int t = threadIdx.x;
    const int idx = (int)(rowmin[m] & 0xFFFFFFFFull);
    const float* q = cb + ((size_t)qi * C_ + idx) * D_;
    float* z = resid + (size_t)m * D_;

    float z0 = z[t], z1 = z[t + 256];
    float q0 = q[t], q1 = q[t + 256];

    float pzz = z0 * z0 + z1 * z1;
    float pqq = q0 * q0 + q1 * q1;
    float pzq = z0 * q0 + z1 * q1;

    __shared__ float s0[256], s1[256], s2[256];
    s0[t] = pzz; s1[t] = pqq; s2[t] = pzq;
    __syncthreads();
    for (int st = 128; st > 0; st >>= 1) {
        if (t < st) { s0[t] += s0[t + st]; s1[t] += s1[t + st]; s2[t] += s2[t + st]; }
        __syncthreads();
    }

    __shared__ float coef[2];
    if (t == 0) {
        float zz = s0[0], qq = s1[0], zq = s2[0];
        float nz = sqrtf(zz), nq = sqrtf(qq);
        float inz = 1.f / (nz + EPS);
        float inq = 1.f / (nq + EPS);
        // w = (z_norm + q_norm)/||.||; e = z - 2(z.w)w + 2(z.z_norm)q_norm
        float ns  = sqrtf(zz * inz * inz + 2.f * zq * inz * inq + qq * inq * inq);
        float zzn = zz * inz;                 // z . z_norm
        float zqn = zq * inq;                 // z . q_norm
        float zw  = (zzn + zqn) / ns;         // z . w
        float alpha = 1.f - 2.f * zw / ns * inz;
        float beta  = (2.f * zzn - 2.f * zw / ns) * inq;
        float scale = nq * inz;               // ||q|| / (||z|| + eps)
        coef[0] = alpha * scale;
        coef[1] = beta * scale;
        loss_rows[m] = zz - 2.f * zq + qq;    // sum((q - z)^2) for this row
        out_idx[(size_t)m * Q_ + qi] = (float)idx;
    }
    __syncthreads();

    float a = coef[0], b = coef[1];
    float quant0 = a * z0 + b * q0;
    float quant1 = a * z1 + b * q1;

    outq[(size_t)m * D_ + t]       += quant0;
    outq[(size_t)m * D_ + t + 256] += quant1;

    float r0 = z0 - quant0;
    float r1 = z1 - quant1;
    z[t] = r0; z[t + 256] = r1;

    _Float16 h0 = (_Float16)r0, h1 = (_Float16)r1;
    rhi[(size_t)m * D_ + t]       = h0;
    rhi[(size_t)m * D_ + t + 256] = h1;
    rlo[(size_t)m * D_ + t]       = (_Float16)(r0 - (float)h0);
    rlo[(size_t)m * D_ + t + 256] = (_Float16)(r1 - (float)h1);
}

// ---------------------------------------------------------------------------
// kernel 3: deterministic loss reduction (fixed order, double accumulator)
// ---------------------------------------------------------------------------
__global__ void k_loss_reduce(const float* __restrict__ loss_rows,
                              float* __restrict__ out_loss, int qi) {
    __shared__ double s[256];
    double acc = 0.0;
    for (int i = threadIdx.x; i < M_; i += 256) acc += (double)loss_rows[i];
    s[threadIdx.x] = acc;
    __syncthreads();
    for (int st = 128; st > 0; st >>= 1) {
        if (threadIdx.x < st) s[threadIdx.x] += s[threadIdx.x + st];
        __syncthreads();
    }
    if (threadIdx.x == 0)
        out_loss[qi] = (float)(s[0] / (double)((size_t)M_ * D_));
}

// ---------------------------------------------------------------------------
// launch
// ---------------------------------------------------------------------------
extern "C" void kernel_launch(void* const* d_in, const int* in_sizes, int n_in,
                              void* d_out, int out_size, void* d_ws, size_t ws_size,
                              hipStream_t stream) {
    const float* x  = (const float*)d_in[0];   // [16, 2048, 512]
    const float* cb = (const float*)d_in[1];   // [8, 1024, 512]

    float* outq     = (float*)d_out;                       // [M_, D_]
    float* out_idx  = outq + (size_t)M_ * D_;              // [M_, Q_]
    float* out_loss = out_idx + (size_t)M_ * Q_;           // [Q_]

    // workspace carve (256B aligned chunks)
    char* ws = (char*)d_ws;
    auto carve = [&](size_t bytes) {
        char* p = ws;
        ws += (bytes + 255) & ~(size_t)255;
        return p;
    };
    float*              resid     = (float*)carve((size_t)M_ * D_ * sizeof(float));
    _Float16*           rhi       = (_Float16*)carve((size_t)M_ * D_ * sizeof(_Float16));
    _Float16*           rlo       = (_Float16*)carve((size_t)M_ * D_ * sizeof(_Float16));
    _Float16*           cbhi      = (_Float16*)carve((size_t)Q_ * C_ * D_ * sizeof(_Float16));
    _Float16*           cblo      = (_Float16*)carve((size_t)Q_ * C_ * D_ * sizeof(_Float16));
    float*              csq       = (float*)carve((size_t)Q_ * C_ * sizeof(float));
    unsigned long long* rowmin    = (unsigned long long*)carve((size_t)M_ * 8);
    float*              loss_rows = (float*)carve((size_t)M_ * sizeof(float));

    k_init_cb<<<Q_ * C_, 256, 0, stream>>>(cb, cbhi, cblo, csq);
    k_init_x<<<4096, 256, 0, stream>>>(x, resid, rhi, rlo, outq);

    for (int qi = 0; qi < Q_; ++qi) {
        k_reset_argmin<<<M_ / 256, 256, 0, stream>>>(rowmin);
        k_dist_argmin<<<4096, 128, 0, stream>>>(rhi, rlo, cbhi, cblo, csq, rowmin, qi);
        k_rotate_update<<<M_, 256, 0, stream>>>(cb, rowmin, resid, rhi, rlo, outq,
                                                loss_rows, out_idx, qi);
        k_loss_reduce<<<1, 256, 0, stream>>>(loss_rows, out_loss, qi);
    }
}
